// SelfAttention_84396107366867
// MI455X (gfx1250) — matrix operand order repaired and checked by reading, not compiled, not run on hardware
//
#include <hip/hip_runtime.h>
#include <hip/hip_bf16.h>

#define BEV    128
#define EMBED  128
#define HEADS  8
#define POINTS 4
#define HD     16
#define NQ     (BEV * BEV)      // 16384 = 2^14
#define BS     4
#define NROWS  (BS * NQ)        // 65536

#define LDS_STRIDE 136          // 128 halves + 8-half pad -> 272B row stride, bank-conflict-free

typedef __attribute__((ext_vector_type(16))) _Float16 v16h;
typedef __attribute__((ext_vector_type(8)))  _Float16 v8h;
typedef __attribute__((ext_vector_type(8)))  float    v8f;

// ---------------------------------------------------------------------------
// WMMA fragment loaders (wave32, v_wmma_f32_16x16x32_f16 layouts, ISA 7.12.2)
// ---------------------------------------------------------------------------
// A: 16x32 f16 from row-major f16 source (two contiguous 16B loads per lane).
__device__ __forceinline__ v16h load_a_frag_f16(const _Float16* A, int lda, int lane, int k0) {
    int m  = lane & 15;
    int kh = (lane >> 4) << 3;
    const _Float16* p = A + (size_t)m * lda + (k0 + kh);
    v8h lo = *(const v8h*)(p);
    v8h hi = *(const v8h*)(p + 16);
    v16h a;
#pragma unroll
    for (int i = 0; i < 8; ++i) { a[i] = lo[i]; a[8 + i] = hi[i]; }
    return a;
}

// A: 16x32 from row-major fp32 source, converted to f16 in-register
// (b128 loads + v_cvt_pk_f16_f32; avoids a separate conversion pass).
__device__ __forceinline__ v16h load_a_frag_f32(const float* A, int lda, int lane, int k0) {
    int m  = lane & 15;
    int kh = (lane >> 4) << 3;
    const float* p = A + (size_t)m * lda + (k0 + kh);
    v16h a;
#pragma unroll
    for (int i = 0; i < 8; ++i) {
        a[i]     = (_Float16)p[i];
        a[8 + i] = (_Float16)p[16 + i];
    }
    return a;
}

// B: 32x16 f16 fragment from padded N-major LDS image: smem[n*LDS_STRIDE + k].
// lane n = lane&15, 8 K at k0 + (lane>>4)*8 and 8 K at +16 -> two 16B ds reads.
// Row stride 272B = 68 banks => lanes 0..15 hit disjoint bank groups.
__device__ __forceinline__ v16h lds_b_frag(const _Float16* smem, int lane, int ntile, int k0) {
    int n  = lane & 15;
    int kb = (lane >> 4) << 3;
    const _Float16* p = smem + (size_t)(ntile * 16 + n) * LDS_STRIDE + (k0 + kb);
    v8h lo = *(const v8h*)(p);
    v8h hi = *(const v8h*)(p + 16);
    v16h b;
#pragma unroll
    for (int i = 0; i < 8; ++i) { b[i] = lo[i]; b[8 + i] = hi[i]; }
    return b;
}

// ---------------------------------------------------------------------------
// Weight transpose (K,N) fp32 -> (N,K) f16, one-time tiny kernels
// ---------------------------------------------------------------------------
__global__ void k_transpose_w(const float* __restrict__ W, _Float16* __restrict__ Wt, int K, int N) {
    int i = blockIdx.x * blockDim.x + threadIdx.x;
    if (i < K * N) {
        int k = i / N, n = i - k * N;
        Wt[(size_t)n * K + k] = (_Float16)W[(size_t)k * N + n];
    }
}

// ---------------------------------------------------------------------------
// Fused projection GEMM: 14 N-tiles over K=128
//   tiles 0..7  : value = query @ W_val + b_val   -> value[b][head][q][hd]
//   tiles 8..11 : off   = query @ W_off + b_off   -> offattn[row][0..63]
//   tiles 12..13: attn  = query @ W_attn + b_attn -> offattn[row][64..95]
// Block = 256 threads (8 waves), 128 rows per block. B staged in LDS,
// B-fragments double-buffered so ds latency pipelines under the XDL wmma.
// ---------------------------------------------------------------------------
__global__ void __launch_bounds__(256) k_proj_fused(
    const float* __restrict__ query,
    const _Float16* __restrict__ WtV, const _Float16* __restrict__ WtO,
    const _Float16* __restrict__ WtA,
    const float* __restrict__ b_val, const float* __restrict__ b_off,
    const float* __restrict__ b_attn,
    float* __restrict__ value, float* __restrict__ offattn)
{
    // 224 weight rows (128 value + 64 off + 32 attn), padded stride
    __shared__ _Float16 smem[224 * LDS_STRIDE];

    // cooperative stage: 224 rows * 16 chunks of 8 halves = 3584 chunks, 14/thread
#pragma unroll
    for (int it = 0; it < 14; ++it) {
        int c   = it * 256 + threadIdx.x;
        int row = c >> 4;
        int ko  = (c & 15) << 3;
        const _Float16* src;
        int srow = row;
        if (row < 128)      { src = WtV; }
        else if (row < 192) { src = WtO; srow = row - 128; }
        else                { src = WtA; srow = row - 192; }
        *(v8h*)(smem + (size_t)row * LDS_STRIDE + ko) =
            *(const v8h*)(src + (size_t)srow * EMBED + ko);
    }
    __syncthreads();

    int lane = threadIdx.x & 31;
    int wave = threadIdx.x >> 5;
    int rowBase = blockIdx.x * 128 + wave * 16;
    const float* Aq = query + (size_t)rowBase * EMBED;

    v8f zero = {};
    v8f acc[14];
#pragma unroll
    for (int t = 0; t < 14; ++t) acc[t] = zero;

#pragma unroll
    for (int k0 = 0; k0 < EMBED; k0 += 32) {
        v16h a = load_a_frag_f32(Aq, EMBED, lane, k0);
        // double-buffered B fragments: issue load t+1 before consuming t
        v16h bcur = lds_b_frag(smem, lane, 0, k0);
#pragma unroll
        for (int t = 0; t < 14; ++t) {
            v16h bnext = (t < 13) ? lds_b_frag(smem, lane, t + 1, k0) : bcur;
            acc[t] = __builtin_amdgcn_wmma_f32_16x16x32_f16(
                false, a, false, bcur, (short)0, acc[t], false, false);
            bcur = bnext;
        }
    }

    int d    = lane & 15;
    int rsel = (lane >> 4) << 3;

    // value tiles: n-tile t == head t (HD == 16), layout (b, head, q, hd)
#pragma unroll
    for (int t = 0; t < 8; ++t) {
        float bias = b_val[t * 16 + d];
#pragma unroll
        for (int r = 0; r < 8; ++r) {
            int grow = rowBase + r + rsel;
            int b = grow >> 14;
            int q = grow & (NQ - 1);
            value[((size_t)(b * HEADS + t) * NQ + q) * HD + d] = acc[t][r] + bias;
        }
    }
    // off/attn tiles: cols 0..95 of offattn
#pragma unroll
    for (int t = 8; t < 14; ++t) {
        int col = (t - 8) * 16 + d;
        float bias = (col < 64) ? b_off[col] : b_attn[col - 64];
#pragma unroll
        for (int r = 0; r < 8; ++r) {
            int grow = rowBase + r + rsel;
            offattn[(size_t)grow * 96 + col] = acc[t][r] + bias;
        }
    }
}

// ---------------------------------------------------------------------------
// Sampling: softmax(4) + bilinear gather; one thread per (b,q,h,d).
// d -> lane: each corner read is a coalesced 64B value row (L2-resident map).
// ---------------------------------------------------------------------------
__device__ __forceinline__ float vsamp(const float* vbase, int x, int y) {
    if ((unsigned)x < (unsigned)BEV && (unsigned)y < (unsigned)BEV)
        return vbase[(size_t)(y * BEV + x) * HD];
    return 0.0f;
}

__global__ void __launch_bounds__(256) k_msda_sample(
    const float* __restrict__ value, const float* __restrict__ offattn,
    _Float16* __restrict__ sampled)
{
    int t  = blockIdx.x * blockDim.x + threadIdx.x;  // < BS*NQ*HEADS*HD
    int d  = t & 15;
    int h  = (t >> 4) & 7;
    int bq = t >> 7;                 // b*NQ + q
    int q  = bq & (NQ - 1);
    int b  = bq >> 14;

    const float* oa = offattn + (size_t)bq * 96;

    float l0 = oa[64 + h * 4 + 0], l1 = oa[64 + h * 4 + 1];
    float l2 = oa[64 + h * 4 + 2], l3 = oa[64 + h * 4 + 3];
    float mx = fmaxf(fmaxf(l0, l1), fmaxf(l2, l3));
    float e[4];
    e[0] = __expf(l0 - mx); e[1] = __expf(l1 - mx);
    e[2] = __expf(l2 - mx); e[3] = __expf(l3 - mx);
    float inv = 1.0f / (e[0] + e[1] + e[2] + e[3]);

    // px = ref_x*W + off_x - 0.5 ; ref = linspace(0,1,BEV)
    const float scale = (float)BEV / (float)(BEV - 1);
    float fx = (float)(q & (BEV - 1)) * scale - 0.5f;
    float fy = (float)(q >> 7)        * scale - 0.5f;

    const float* vbase = value + (size_t)(b * HEADS + h) * NQ * HD + d;

    float acc = 0.0f;
#pragma unroll
    for (int p = 0; p < POINTS; ++p) {
        float aw = e[p] * inv;
        float px = fx + oa[(h * 4 + p) * 2 + 0];
        float py = fy + oa[(h * 4 + p) * 2 + 1];
        float x0f = floorf(px), y0f = floorf(py);
        float wx = px - x0f, wy = py - y0f;
        int x0 = (int)x0f, y0 = (int)y0f;
        float s00 = vsamp(vbase, x0,     y0);
        float s01 = vsamp(vbase, x0 + 1, y0);
        float s10 = vsamp(vbase, x0,     y0 + 1);
        float s11 = vsamp(vbase, x0 + 1, y0 + 1);
        float bil = s00 * (1.0f - wx) * (1.0f - wy) + s01 * wx * (1.0f - wy)
                  + s10 * (1.0f - wx) * wy          + s11 * wx * wy;
        acc += aw * bil;
    }
    sampled[(size_t)bq * EMBED + h * 16 + d] = (_Float16)acc;
}

// ---------------------------------------------------------------------------
// Output GEMM: out = sampled @ W_out + b_out + 2*query (W_out staged in LDS)
// ---------------------------------------------------------------------------
__global__ void __launch_bounds__(256) k_gemm_out(
    const _Float16* __restrict__ Sh, const _Float16* __restrict__ WtU,
    const float* __restrict__ b_out, const float* __restrict__ query,
    float* __restrict__ out)
{
    __shared__ _Float16 smem[128 * LDS_STRIDE];

    // stage: 128 rows * 16 chunks = 2048 chunks, 8/thread
#pragma unroll
    for (int it = 0; it < 8; ++it) {
        int c   = it * 256 + threadIdx.x;
        int row = c >> 4;
        int ko  = (c & 15) << 3;
        *(v8h*)(smem + (size_t)row * LDS_STRIDE + ko) =
            *(const v8h*)(WtU + (size_t)row * EMBED + ko);
    }
    __syncthreads();

    int lane = threadIdx.x & 31;
    int wave = threadIdx.x >> 5;
    int rowBase = blockIdx.x * 128 + wave * 16;
    const _Float16* A = Sh + (size_t)rowBase * EMBED;

    v8f zero = {};
    v8f acc[8];
#pragma unroll
    for (int t = 0; t < 8; ++t) acc[t] = zero;

#pragma unroll
    for (int k0 = 0; k0 < EMBED; k0 += 32) {
        v16h a = load_a_frag_f16(A, EMBED, lane, k0);
        v16h bcur = lds_b_frag(smem, lane, 0, k0);
#pragma unroll
        for (int t = 0; t < 8; ++t) {
            v16h bnext = (t < 7) ? lds_b_frag(smem, lane, t + 1, k0) : bcur;
            acc[t] = __builtin_amdgcn_wmma_f32_16x16x32_f16(
                false, a, false, bcur, (short)0, acc[t], false, false);
            bcur = bnext;
        }
    }

    int d    = lane & 15;
    int rsel = (lane >> 4) << 3;
#pragma unroll
    for (int t = 0; t < 8; ++t) {
        int col = t * 16 + d;
        float bias = b_out[col];
#pragma unroll
        for (int r = 0; r < 8; ++r) {
            int grow = rowBase + r + rsel;
            size_t idx = (size_t)grow * EMBED + col;
            out[idx] = acc[t][r] + bias + 2.0f * query[idx];
        }
    }
}

// ---------------------------------------------------------------------------
// Host launcher
// ---------------------------------------------------------------------------
extern "C" void kernel_launch(void* const* d_in, const int* in_sizes, int n_in,
                              void* d_out, int out_size, void* d_ws, size_t ws_size,
                              hipStream_t stream) {
    const float* query  = (const float*)d_in[0];
    const float* W_off  = (const float*)d_in[1];
    const float* b_off  = (const float*)d_in[2];
    const float* W_attn = (const float*)d_in[3];
    const float* b_attn = (const float*)d_in[4];
    const float* W_val  = (const float*)d_in[5];
    const float* b_val  = (const float*)d_in[6];
    const float* W_out  = (const float*)d_in[7];
    const float* b_out  = (const float*)d_in[8];
    float* out = (float*)d_out;

    // workspace carve-up (256B aligned)
    char* ws = (char*)d_ws;
    size_t off = 0;
    auto carve = [&](size_t bytes) -> void* {
        void* p = ws + off;
        off = (off + bytes + 255) & ~(size_t)255;
        return p;
    };
    _Float16* WtV     = (_Float16*)carve((size_t)EMBED * EMBED * sizeof(_Float16));
    _Float16* WtO     = (_Float16*)carve((size_t)EMBED * 64    * sizeof(_Float16));
    _Float16* WtA     = (_Float16*)carve((size_t)EMBED * 32    * sizeof(_Float16));
    _Float16* WtU     = (_Float16*)carve((size_t)EMBED * EMBED * sizeof(_Float16));
    float*    value   = (float*)   carve((size_t)NROWS * EMBED * sizeof(float));
    float*    offattn = (float*)   carve((size_t)NROWS * 96    * sizeof(float));
    _Float16* Sh      = (_Float16*)carve((size_t)NROWS * EMBED * sizeof(_Float16));
    (void)ws_size; (void)in_sizes; (void)n_in; (void)out_size;

    // one-time tiny weight transposes (N-major f16), L2-resident thereafter
    k_transpose_w<<<(EMBED * EMBED + 255) / 256, 256, 0, stream>>>(W_val,  WtV, EMBED, EMBED);
    k_transpose_w<<<(EMBED * 64   + 255) / 256, 256, 0, stream>>>(W_off,  WtO, EMBED, 64);
    k_transpose_w<<<(EMBED * 32   + 255) / 256, 256, 0, stream>>>(W_attn, WtA, EMBED, 32);
    k_transpose_w<<<(EMBED * EMBED + 255) / 256, 256, 0, stream>>>(W_out,  WtU, EMBED, EMBED);

    // pipeline
    k_proj_fused <<<NROWS / 128, 256, 0, stream>>>(query, WtV, WtO, WtA,
                                                   b_val, b_off, b_attn, value, offattn);
    k_msda_sample<<<(NROWS * EMBED) / 256, 256, 0, stream>>>(value, offattn, Sh);
    k_gemm_out   <<<NROWS / 128, 256, 0, stream>>>(Sh, WtU, b_out, query, out);
}
